// gcn_model_75230647156742
// MI455X (gfx1250) — matrix operand order repaired and checked by reading, not compiled, not run on hardware
//
#include <hip/hip_runtime.h>

typedef __attribute__((ext_vector_type(16))) _Float16 v16h;
typedef __attribute__((ext_vector_type(8)))  float    v8f;
typedef __attribute__((ext_vector_type(4)))  int      v4i;

#define LDS_AS __attribute__((address_space(3)))
#define GLB_AS __attribute__((address_space(1)))

typedef GLB_AS v4i* gv4i_p;
typedef LDS_AS v4i* lv4i_p;

// ---------------------------------------------------------------------------
// Degree / normalization
// ---------------------------------------------------------------------------
__global__ void k_deg_init(float* __restrict__ dinv, int n) {
    int i = blockIdx.x * blockDim.x + threadIdx.x;
    if (i < n) dinv[i] = 1.0f;                       // self-loop contributes 1
}

__global__ void k_deg_scatter(const long long* __restrict__ ei,
                              float* __restrict__ dinv, int E) {
    int e = blockIdx.x * blockDim.x + threadIdx.x;
    if (e < E) unsafeAtomicAdd(&dinv[(int)ei[(size_t)E + e]], 1.0f);
}

__global__ void k_dinv(float* __restrict__ dinv, int n) {
    int i = blockIdx.x * blockDim.x + threadIdx.x;
    if (i < n) dinv[i] = __frsqrt_rn(dinv[i]);       // deg >= 1 always
}

// ---------------------------------------------------------------------------
// GCN node transforms (fused: prev bias+ReLU, GEMV, self-loop init of agg)
// ---------------------------------------------------------------------------
__global__ void k_xform1(const float* __restrict__ x, const float* __restrict__ W,
                         const float* __restrict__ dinv, float* __restrict__ ht,
                         float* __restrict__ agg, int n) {
    int i = blockIdx.x * blockDim.x + threadIdx.x;
    if (i >= n) return;
    float xi[9];
#pragma unroll
    for (int k = 0; k < 9; ++k) xi[k] = x[(size_t)i * 9 + k];
    float d2 = dinv[i]; d2 *= d2;
#pragma unroll
    for (int c = 0; c < 16; ++c) {
        float s = 0.f;
#pragma unroll
        for (int k = 0; k < 9; ++k) s = fmaf(xi[k], W[k * 16 + c], s);
        ht[(size_t)i * 16 + c]  = s;
        agg[(size_t)i * 16 + c] = s * d2;            // self-loop message
    }
}

__global__ void k_xform16(const float* __restrict__ b_prev, const float* __restrict__ W,
                          const float* __restrict__ dinv, float* __restrict__ ht,
                          float* __restrict__ agg, int n) {
    int i = blockIdx.x * blockDim.x + threadIdx.x;
    if (i >= n) return;
    float h[16];
#pragma unroll
    for (int k = 0; k < 16; ++k)
        h[k] = fmaxf(agg[(size_t)i * 16 + k] + b_prev[k], 0.f);
    float d2 = dinv[i]; d2 *= d2;
#pragma unroll
    for (int c = 0; c < 16; ++c) {
        float s = 0.f;
#pragma unroll
        for (int k = 0; k < 16; ++k) s = fmaf(h[k], W[k * 16 + c], s);
        ht[(size_t)i * 16 + c]  = s;
        agg[(size_t)i * 16 + c] = s * d2;
    }
}

// ---------------------------------------------------------------------------
// Edge scatter: agg[dst] += ht[src] * dinv[src] * dinv[dst]
// ---------------------------------------------------------------------------
__global__ void k_edge_scatter(const long long* __restrict__ ei,
                               const float* __restrict__ dinv,
                               const float* __restrict__ ht,
                               float* __restrict__ agg, int E) {
    int e = blockIdx.x * blockDim.x + threadIdx.x;
    if (e >= E) return;
    long long s = ei[e];
    long long d = ei[(size_t)E + e];
    float nrm = dinv[s] * dinv[d];
    const float4* hs = (const float4*)(ht + s * 16);
    float* ad = agg + d * 16;
#pragma unroll
    for (int q = 0; q < 4; ++q) {
        float4 v = hs[q];
        unsafeAtomicAdd(ad + q * 4 + 0, v.x * nrm);
        unsafeAtomicAdd(ad + q * 4 + 1, v.y * nrm);
        unsafeAtomicAdd(ad + q * 4 + 2, v.z * nrm);
        unsafeAtomicAdd(ad + q * 4 + 3, v.w * nrm);
    }
}

// ---------------------------------------------------------------------------
// Pre-pack MLP weights into f16 WMMA B-fragment layout (contiguous wf2|wf3).
// B order used here (contiguous k, 32x16 per K-chunk): lane<16 -> K=j, lane>=16 -> K=16+j;
// N = t*16 + (lane&15).  Fragment idx = ((t*KCH + kc)*32 + lane)*16 + j.
// ---------------------------------------------------------------------------
__global__ void k_prep_w(const float* __restrict__ Wl2, const float* __restrict__ Wl3,
                         _Float16* __restrict__ wf2, _Float16* __restrict__ wf3) {
    int tid = blockIdx.x * blockDim.x + threadIdx.x;
    int stride = gridDim.x * blockDim.x;
    for (int idx = tid; idx < 8 * 512; idx += stride) {        // Wl2: 8 ntiles, 1 kchunk
        int j = idx & 15, lane = (idx >> 4) & 31, t = idx >> 9;
        int k = ((lane >= 16) ? 16 : 0) + j;
        int nn = t * 16 + (lane & 15);
        wf2[idx] = (_Float16)((k < 25) ? Wl2[k * 128 + nn] : 0.f);  // pad K 25->32
    }
    for (int idx = tid; idx < 32 * 512; idx += stride) {       // Wl3: 8 ntiles, 4 kchunks
        int j = idx & 15, lane = (idx >> 4) & 31, kc = (idx >> 9) & 3, t = idx >> 11;
        int k = kc * 32 + ((lane >= 16) ? 16 : 0) + j;
        int nn = t * 16 + (lane & 15);
        wf3[idx] = (_Float16)Wl3[k * 128 + nn];
    }
}

// ---------------------------------------------------------------------------
// Fused MLP head: [x | relu(agg3+b3)] (25) -> 128 -> 128 -> 1, WMMA f16.
// 128 threads = 4 waves / block; one wave per 16-node tile.
// Weights staged once per block into LDS via async global->LDS copy.
// ---------------------------------------------------------------------------
#define MLP_WAVES 4
#define WFRAG_HALVES (8 * 512 + 32 * 512)          // 20480 halves = 40 KB

__global__ void __launch_bounds__(128) k_mlp(
    const float* __restrict__ x, const float* __restrict__ agg3,
    const float* __restrict__ b3,
    const _Float16* __restrict__ wfrag,            // wf2 (4096) | wf3 (16384)
    const float* __restrict__ bl2,
    const float* __restrict__ bl3,
    const float* __restrict__ wl4, const float* __restrict__ bl4,
    float* __restrict__ out, int n)
{
    __shared__ _Float16 wls[WFRAG_HALVES];                 // 40 KB packed weights
    __shared__ _Float16 h2s[MLP_WAVES][16][136];           // padded vs. bank conflicts

    const int tid   = threadIdx.x;
    const int lane  = tid & 31;
    const int w     = tid >> 5;
    const int m     = lane & 15;           // row within tile / N within C-frag
    const int khi   = (lane >> 4) * 8;     // (0 or 8); A-frag K offset is 2*khi (0 or 16)
    const int rbase = (lane >> 4) * 8;     // C-frag row base (0 or 8)
    const int base  = (blockIdx.x * MLP_WAVES + w) * 16;
    const int node  = base + m;
    const bool valid = node < n;

    // ---- stage all weight fragments into LDS (async copy, 16 B / lane / op) ----
#if __has_builtin(__builtin_amdgcn_global_load_async_to_lds_b128)
#pragma unroll
    for (int it = tid; it < WFRAG_HALVES / 8; it += 128)   // 2560 x b128
        __builtin_amdgcn_global_load_async_to_lds_b128(
            (gv4i_p)(wfrag + (size_t)it * 8),
            (lv4i_p)(wls + (size_t)it * 8), 0, 0);
#if __has_builtin(__builtin_amdgcn_s_wait_asynccnt)
    __builtin_amdgcn_s_wait_asynccnt(0);
#else
    asm volatile("s_wait_asynccnt 0" ::: "memory");
#endif
#else
    // fallback: vectorized load + LDS store
    for (int it = tid; it < WFRAG_HALVES / 8; it += 128)
        ((float4*)wls)[it] = ((const float4*)wfrag)[it];
#endif
    __syncthreads();

    // ---- layer-2 A fragment: 16x32 f16, K = [x(9) | relu(agg3+b3)(16) | 0-pad] ----
    v16h a2;
#pragma unroll
    for (int j = 0; j < 16; ++j) {
        int k = j + 2 * khi;   // A layout: contiguous, k = 16*(lane/16) + j
        float v = 0.f;
        if (valid) {
            if (k < 9)        v = x[(size_t)node * 9 + k];
            else if (k < 25) { int c = k - 9;
                               v = fmaxf(agg3[(size_t)node * 16 + c] + b3[c], 0.f); }
        }
        a2[j] = (_Float16)v;
    }

    const v16h* bf2 = (const v16h*)wls;            // 256 fragments
    const v16h* bf3 = bf2 + 8 * 32;                // 1024 fragments

    // ---- layer 2: (25->128) + bias + relu, result to LDS as f16 ----
#pragma unroll
    for (int t = 0; t < 8; ++t) {
        v8f acc = {0.f, 0.f, 0.f, 0.f, 0.f, 0.f, 0.f, 0.f};
        acc = __builtin_amdgcn_wmma_f32_16x16x32_f16(false, a2, false,
                  bf2[t * 32 + lane], (short)0, acc, false, false);
        float bb = bl2[t * 16 + m];
#pragma unroll
        for (int r = 0; r < 8; ++r) {
            float v = fmaxf(acc[r] + bb, 0.f);
            h2s[w][rbase + r][t * 16 + m] = (_Float16)v;   // C-layout -> [row][col]
        }
    }
    // per-wave private LDS region: same-wave DS ordering suffices (no barrier)

    // ---- layer-3 A fragments (16x128 f16 = 4 K-chunks) from LDS ----
    v16h a3[4];
#pragma unroll
    for (int kc = 0; kc < 4; ++kc)
#pragma unroll
        for (int j = 0; j < 16; ++j) {
            int k = kc * 32 + j + 2 * khi;
            a3[kc][j] = h2s[w][m][k];
        }

    // ---- layer 3 (128->128) + bias + relu, fused layer-4 dot (128->1) ----
    float partial[8] = {0.f, 0.f, 0.f, 0.f, 0.f, 0.f, 0.f, 0.f};
#pragma unroll
    for (int t = 0; t < 8; ++t) {
        v8f acc = {0.f, 0.f, 0.f, 0.f, 0.f, 0.f, 0.f, 0.f};
#pragma unroll
        for (int kc = 0; kc < 4; ++kc)
            acc = __builtin_amdgcn_wmma_f32_16x16x32_f16(false, a3[kc], false,
                      bf3[(t * 4 + kc) * 32 + lane], (short)0, acc, false, false);
        float bb = bl3[t * 16 + m];
        float w4 = wl4[t * 16 + m];          // this lane's column of Wl4
#pragma unroll
        for (int r = 0; r < 8; ++r)
            partial[r] = fmaf(fmaxf(acc[r] + bb, 0.f), w4, partial[r]);
    }

    // reduce over the 16 lanes sharing each output row (stays in half-wave)
#pragma unroll
    for (int mask = 1; mask <= 8; mask <<= 1)
#pragma unroll
        for (int r = 0; r < 8; ++r)
            partial[r] += __shfl_xor(partial[r], mask, 32);

    if (m == 0) {                            // lanes 0 (rows 0-7) and 16 (rows 8-15)
        float bb4 = bl4[0];
#pragma unroll
        for (int r = 0; r < 8; ++r) {
            int o = base + rbase + r;
            if (o < n) out[o] = partial[r] + bb4;
        }
    }
}

// ---------------------------------------------------------------------------
extern "C" void kernel_launch(void* const* d_in, const int* in_sizes, int n_in,
                              void* d_out, int out_size, void* d_ws, size_t ws_size,
                              hipStream_t stream) {
    const float*     x   = (const float*)d_in[0];
    const long long* ei  = (const long long*)d_in[1];   // int64 edge_index (2,E)
    const float* W1  = (const float*)d_in[2];
    const float* b1  = (const float*)d_in[3];
    const float* W2  = (const float*)d_in[4];
    const float* b2  = (const float*)d_in[5];
    const float* W3  = (const float*)d_in[6];
    const float* b3  = (const float*)d_in[7];
    const float* Wl2 = (const float*)d_in[8];
    const float* bl2 = (const float*)d_in[9];
    const float* Wl3 = (const float*)d_in[10];
    const float* bl3 = (const float*)d_in[11];
    const float* Wl4 = (const float*)d_in[12];
    const float* bl4 = (const float*)d_in[13];
    float* out = (float*)d_out;

    const int N = in_sizes[0] / 9;
    const int E = in_sizes[1] / 2;

    // workspace layout (floats): dinv[N] | ht[16N] | agg[16N] | f16 weight frags
    float* f_ws = (float*)d_ws;
    size_t off = 0;
    float* dinv = f_ws + off; off += (size_t)N;
    float* ht   = f_ws + off; off += (size_t)N * 16;
    float* agg  = f_ws + off; off += (size_t)N * 16;
    off = (off + 7) & ~(size_t)7;                       // 32-B align for v16h
    _Float16* wf2 = (_Float16*)(f_ws + off);
    _Float16* wf3 = wf2 + 8 * 512;                      // contiguous after wf2

    dim3 blk(256);
    dim3 gN((N + 255) / 256), gE((E + 255) / 256);

    k_deg_init   <<<gN, blk, 0, stream>>>(dinv, N);
    k_deg_scatter<<<gE, blk, 0, stream>>>(ei, dinv, E);
    k_dinv       <<<gN, blk, 0, stream>>>(dinv, N);

    k_xform1      <<<gN, blk, 0, stream>>>(x, W1, dinv, ht, agg, N);
    k_edge_scatter<<<gE, blk, 0, stream>>>(ei, dinv, ht, agg, E);
    k_xform16     <<<gN, blk, 0, stream>>>(b1, W2, dinv, ht, agg, N);
    k_edge_scatter<<<gE, blk, 0, stream>>>(ei, dinv, ht, agg, E);
    k_xform16     <<<gN, blk, 0, stream>>>(b2, W3, dinv, ht, agg, N);
    k_edge_scatter<<<gE, blk, 0, stream>>>(ei, dinv, ht, agg, E);

    k_prep_w<<<dim3(64), blk, 0, stream>>>(Wl2, Wl3, wf2, wf3);

    dim3 gM((N + MLP_WAVES * 16 - 1) / (MLP_WAVES * 16));   // 64 nodes / block
    k_mlp<<<gM, dim3(128), 0, stream>>>(x, agg, b3, wf2, bl2, bl3, Wl4, bl4, out, N);
}
